// AgentGNN_26723286516031
// MI455X (gfx1250) — compile-verified, matrix operand order checked by reading, not run on hardware
//
#include <hip/hip_runtime.h>
#include <hip/hip_bf16.h>
#include <math.h>

typedef _Float16 half_t;
typedef __attribute__((ext_vector_type(16))) _Float16 v16h;
typedef __attribute__((ext_vector_type(8)))  float    v8f;

static constexpr int Bsz   = 32;
static constexpr int Nn    = 2048;
static constexpr int Ii    = 64;
static constexpr int Dd    = 128;
static constexpr int Aa    = 64;
static constexpr int ROWS  = Bsz * Nn;   // 65536
static constexpr int AROWS = Bsz * Aa;   // 2048
static constexpr int NSTEPS = 8;

// tvec layout: [t_node 448 | t_agent 384 | t_conv 320 | t_glob 128 | t_read 128]
#define TN_OFF 0
#define TA_OFF 448
#define TC_OFF 832
#define TG_OFF 1152
#define TR_OFF 1280
#define TVEC_LEN 1408

// ---------------------------------------------------------------------------
// block reduction helper (broadcasts result, leaves buffer reusable)
// ---------------------------------------------------------------------------
template<int BS>
__device__ __forceinline__ float block_sum(float v, float* red) {
    int tid = threadIdx.x;
    red[tid] = v; __syncthreads();
    #pragma unroll
    for (int o = BS >> 1; o > 0; o >>= 1) {
        if (tid < o) red[tid] += red[tid + o];
        __syncthreads();
    }
    float r = red[0]; __syncthreads();
    return r;
}

// ---------------------------------------------------------------------------
// WMMA fragment load: row-major [rows, K] f16, 16-bit A-matrix layout
// lane L holds row (tile + L%16); K pattern per VGPR pair + 8*(L/16).
// B operands are stored transposed (N x K), so one loader serves both.
// Per-lane bytes are contiguous at p[0..15] and p[32..47] -> two b128 loads.
// ---------------------------------------------------------------------------
__device__ __forceinline__ v16h frag_load(const half_t* __restrict__ base,
                                          int row, int K, int k0, int lane) {
    union { v16h v; unsigned u[8]; } f;
    const half_t* p = base + (size_t)row * K + k0 + ((lane >> 4) << 3);
    #pragma unroll
    for (int i = 0; i < 8; ++i) {
        int kb = (i < 4) ? (i << 1) : (16 + ((i - 4) << 1));
        f.u[i] = *(const unsigned*)(p + kb);
    }
    return f.v;
}

// ---------------------------------------------------------------------------
// Generic batched WMMA GEMM: C[b] = act(alpha * A[b] * Bt[b]^T + bias + beta)
// A: [M,K] f16 row-major, Bt: [N,K] f16 row-major, C: [M,N]
// block = 256 threads = 8 waves arranged 2 (M) x 4 (N).
// Each wave computes a (16*MT) x (16*NT) register tile: A fragments reused
// NT times, B fragments MT times (arithmetic-intensity blocking).
// act: 0=none, 1=leaky relu(0.01)
// ---------------------------------------------------------------------------
template<int MT, int NT>
__global__ __launch_bounds__(256) void gemm_wmma_kernel(
    const half_t* __restrict__ A, const half_t* __restrict__ Bt,
    const float* __restrict__ bias, float* __restrict__ Cf,
    half_t* __restrict__ Ch,
    int M, int N, int K,
    long long sA, long long sB, long long sC,
    float alpha, float beta, int act)
{
    int lane = threadIdx.x & 31;
    int wave = threadIdx.x >> 5;
    int wrow = wave >> 2;                 // 0..1
    int wcol = wave & 3;                  // 0..3
    int tm = blockIdx.y * (32 * MT) + wrow * (16 * MT);
    int tn = blockIdx.x * (64 * NT) + wcol * (16 * NT);
    int b  = blockIdx.z;
    if (tm >= M || tn >= N) return;

    const half_t* Ab = A  + (long long)b * sA;
    const half_t* Bb = Bt + (long long)b * sB;

    v8f c[MT][NT] = {};
    int lrow = lane & 15;
    for (int k0 = 0; k0 < K; k0 += 32) {
        // prefetch streamed A operand one K-chunk ahead (global_prefetch_b8)
        if (k0 + 64 <= K)
            __builtin_prefetch(Ab + (size_t)(tm + lrow) * K + k0 + 32, 0, 1);
        v16h af[MT], bf[NT];
        #pragma unroll
        for (int mi = 0; mi < MT; ++mi)
            af[mi] = frag_load(Ab, tm + mi * 16 + lrow, K, k0, lane);
        #pragma unroll
        for (int ni = 0; ni < NT; ++ni)
            bf[ni] = frag_load(Bb, tn + ni * 16 + lrow, K, k0, lane);
        #pragma unroll
        for (int mi = 0; mi < MT; ++mi)
            #pragma unroll
            for (int ni = 0; ni < NT; ++ni)
                c[mi][ni] = __builtin_amdgcn_wmma_f32_16x16x32_f16(
                    false, af[mi], false, bf[ni], (short)0, c[mi][ni],
                    false, false);
    }

    #pragma unroll
    for (int mi = 0; mi < MT; ++mi) {
        #pragma unroll
        for (int ni = 0; ni < NT; ++ni) {
            int m0 = tm + mi * 16 + 8 * (lane >> 4);
            int n  = tn + ni * 16 + (lane & 15);
            float bs = bias ? bias[n] : 0.0f;
            #pragma unroll
            for (int v = 0; v < 8; ++v) {
                float val = alpha * c[mi][ni][v] + bs + beta;
                if (act == 1) val = (val > 0.f) ? val : 0.01f * val;
                long long idx = (long long)b * sC + (long long)(m0 + v) * N + n;
                if (Cf) Cf[idx] = val;
                if (Ch) Ch[idx] = (half_t)val;
            }
        }
    }
}

// ---------------------------------------------------------------------------
// small utility kernels
// ---------------------------------------------------------------------------
__global__ void wtrans_kernel(const float* __restrict__ w, half_t* __restrict__ wt,
                              int fin, int fout) {
    long long idx = (long long)blockIdx.x * blockDim.x + threadIdx.x;
    long long tot = (long long)fin * fout;
    if (idx >= tot) return;
    int o = (int)(idx / fin), i = (int)(idx % fin);
    wt[(long long)o * fin + i] = (half_t)w[(long long)i * fout + o];
}

__global__ void convert_h_kernel(const float* __restrict__ s, half_t* __restrict__ d,
                                 long long n) {
    long long idx = (long long)blockIdx.x * blockDim.x + threadIdx.x;
    if (idx < n) d[idx] = (half_t)s[idx];
}

__global__ void fillf_kernel(float* p, float v, long long n) {
    long long idx = (long long)blockIdx.x * blockDim.x + threadIdx.x;
    if (idx < n) p[idx] = v;
}

__global__ void init_agent_kernel(const float* __restrict__ tab, float* __restrict__ agent) {
    int idx = blockIdx.x * blockDim.x + threadIdx.x;   // AROWS*Dd
    if (idx >= AROWS * Dd) return;
    int row = idx >> 7, d = idx & 127;
    int a = row & (Aa - 1);
    agent[idx] = tab[a * Dd + d];
}

__global__ __launch_bounds__(256) void init_state_kernel(
    const int* __restrict__ start_pos, float* __restrict__ vis,
    int* pos, int* prev, float* attn_val) {
    int row = blockIdx.x;               // b*A + a
    int b = row >> 6;
    int p = start_pos[b];
    for (int n = threadIdx.x; n < Nn; n += 256)
        vis[(size_t)row * Nn + n] = (n == p) ? 1.f : 0.f;
    if (threadIdx.x == 0) { pos[row] = p; prev[row] = p; attn_val[row] = 1.f; }
}

// ---------------------------------------------------------------------------
// time embedding + all five time projections (one block)
// ---------------------------------------------------------------------------
__global__ __launch_bounds__(512) void time_kernel(
    const float* __restrict__ table, int t,
    const float* w1, const float* b1, const float* w2, const float* b2,
    const float* wn, const float* bn, const float* wa, const float* ba,
    const float* wc, const float* bc, const float* wg, const float* bg,
    const float* wr, const float* br, float* __restrict__ tvec)
{
    __shared__ float te[128], h1[256], h2[256];
    int tid = threadIdx.x;
    if (tid < 128) te[tid] = table[(size_t)t * 128 + tid];
    __syncthreads();
    for (int o = tid; o < 256; o += 512) {
        float acc = b1[o];
        for (int i = 0; i < 128; ++i) acc += te[i] * w1[i * 256 + o];
        h1[o] = (acc > 0.f) ? acc : 0.01f * acc;
    }
    __syncthreads();
    for (int o = tid; o < 256; o += 512) {
        float acc = b2[o];
        for (int i = 0; i < 256; ++i) acc += h1[i] * w2[i * 256 + o];
        h2[o] = (acc > 0.f) ? acc : 0.01f * acc;   // lrelu(te2)
    }
    __syncthreads();
    for (int o = tid; o < 448; o += 512) {
        float acc = bn[o];
        for (int i = 0; i < 256; ++i) acc += h2[i] * wn[i * 448 + o];
        tvec[TN_OFF + o] = acc;
    }
    for (int o = tid; o < 384; o += 512) {
        float acc = ba[o];
        for (int i = 0; i < 256; ++i) acc += h2[i] * wa[i * 384 + o];
        tvec[TA_OFF + o] = acc;
    }
    for (int o = tid; o < 320; o += 512) {
        float acc = bc[o];
        for (int i = 0; i < 256; ++i) acc += h2[i] * wc[i * 320 + o];
        tvec[TC_OFF + o] = acc;
    }
    for (int o = tid; o < 128; o += 512) {
        float acc = bg[o];
        for (int i = 0; i < 256; ++i) acc += h2[i] * wg[i * 128 + o];
        tvec[TG_OFF + o] = acc;
    }
    for (int o = tid; o < 128; o += 512) {
        float acc = br[o];
        for (int i = 0; i < 256; ++i) acc += h2[i] * wr[i * 128 + o];
        tvec[TR_OFF + o] = acc;
    }
}

// ---------------------------------------------------------------------------
// msg = relu(lin(ln(agent, msg_ln), msg_l))     (AROWS blocks x 128)
// ---------------------------------------------------------------------------
__global__ __launch_bounds__(128) void msg_kernel(
    const float* __restrict__ agent, const float* g, const float* bln,
    const float* w, const float* bw, float* __restrict__ msg)
{
    int row = blockIdx.x, tid = threadIdx.x;
    __shared__ float x[128], red[128];
    float v = agent[(size_t)row * 128 + tid];
    float mean = block_sum<128>(v, red) / 128.f;
    float d = v - mean;
    float var = block_sum<128>(d * d, red) / 128.f;
    float rstd = rsqrtf(var + 1e-5f);
    x[tid] = d * rstd * g[tid] + bln[tid];
    __syncthreads();
    float acc = bw[tid];
    for (int i = 0; i < 128; ++i) acc += x[i] * w[i * 128 + tid];
    msg[(size_t)row * 128 + tid] = (acc > 0.f) ? acc : 0.f;
}

__global__ __launch_bounds__(128) void scatter_kernel(
    const float* __restrict__ msg, const float* __restrict__ attn_val,
    const int* __restrict__ pos, float* __restrict__ agg)
{
    int row = blockIdx.x, d = threadIdx.x;
    int b = row >> 6;
    int p = pos[row];
    atomicAdd(&agg[((size_t)b * Nn + p) * 128 + d],
              msg[(size_t)row * 128 + d] * attn_val[row]);
}

// ---------------------------------------------------------------------------
// glob = l2(lrelu(l1(ln(mean_a(agent) + t_glob))))    (Bsz blocks x 128)
// ---------------------------------------------------------------------------
__global__ __launch_bounds__(128) void glob_kernel(
    const float* __restrict__ agent, const float* tvec,
    const float* g, const float* bln,
    const float* w1, const float* b1, const float* w2, const float* b2,
    float* __restrict__ glob)
{
    int b = blockIdx.x, d = threadIdx.x;
    __shared__ float x[128], red[128], hid[256];
    float s = 0.f;
    for (int a = 0; a < Aa; ++a) s += agent[((size_t)(b * Aa + a)) * 128 + d];
    float v = s / (float)Aa + tvec[TG_OFF + d];
    float mean = block_sum<128>(v, red) / 128.f;
    float dv = v - mean;
    float var = block_sum<128>(dv * dv, red) / 128.f;
    float rstd = rsqrtf(var + 1e-5f);
    x[d] = dv * rstd * g[d] + bln[d];
    __syncthreads();
    for (int o = d; o < 256; o += 128) {
        float acc = b1[o];
        for (int i = 0; i < 128; ++i) acc += x[i] * w1[i * 256 + o];
        hid[o] = (acc > 0.f) ? acc : 0.01f * acc;
    }
    __syncthreads();
    float acc = b2[d];
    for (int i = 0; i < 256; ++i) acc += hid[i] * w2[i * 128 + d];
    glob[(size_t)b * 128 + d] = acc;
}

// ---------------------------------------------------------------------------
// n_ln = f16( LN([h|agg|glob|init] + t_node, node_ln) )  (ROWS blocks x 256)
// ---------------------------------------------------------------------------
__global__ __launch_bounds__(256) void ln_node_kernel(
    const float* __restrict__ h, const float* __restrict__ agg,
    const float* __restrict__ glob, const float* __restrict__ init,
    const float* tvec, const float* g, const float* bln,
    half_t* __restrict__ out)
{
    int row = blockIdx.x;
    int b = row >> 11;
    __shared__ float sh[448], red[256];
    for (int i = threadIdx.x; i < 448; i += 256) {
        float v;
        if      (i < 128) v = h[(size_t)row * 128 + i];
        else if (i < 256) v = agg[(size_t)row * 128 + (i - 128)];
        else if (i < 384) v = glob[(size_t)b * 128 + (i - 256)];
        else              v = init[(size_t)row * 64 + (i - 384)];
        sh[i] = v + tvec[TN_OFF + i];
    }
    __syncthreads();
    float s = 0.f;
    for (int i = threadIdx.x; i < 448; i += 256) s += sh[i];
    float mean = block_sum<256>(s, red) / 448.f;
    float s2 = 0.f;
    for (int i = threadIdx.x; i < 448; i += 256) { float d = sh[i] - mean; s2 += d * d; }
    float var = block_sum<256>(s2, red) / 448.f;
    float rstd = rsqrtf(var + 1e-5f);
    for (int i = threadIdx.x; i < 448; i += 256)
        out[(size_t)row * 448 + i] = (half_t)((sh[i] - mean) * rstd * g[i] + bln[i]);
}

__global__ void gate_h_kernel(const float* __restrict__ ug, float* __restrict__ h) {
    long long idx = (long long)blockIdx.x * blockDim.x + threadIdx.x;
    if (idx >= (long long)ROWS * Dd) return;
    long long row = idx >> 7; int d = (int)(idx & 127);
    float u = ug[row * 256 + d];
    float gg = ug[row * 256 + 128 + d];
    float sg = 1.f / (1.f + expf(-gg));
    h[idx] = sg * h[idx] + (1.f - sg) * u;
}

// ---------------------------------------------------------------------------
// conv (gated agent-local conv), AROWS blocks x 256
// ---------------------------------------------------------------------------
__global__ __launch_bounds__(256) void conv_kernel(
    const float* __restrict__ h, const float* __restrict__ agent,
    const float* __restrict__ init, const int* __restrict__ pos,
    const float* tvec, const float* g, const float* bln,
    const float* w1, const float* b1, const float* w2, const float* b2,
    float* __restrict__ convout)
{
    int row = blockIdx.x;
    int b = row >> 6;
    int p = pos[row];
    __shared__ float cin[320], hid[512], red[256];
    const float* hat = h    + ((size_t)b * Nn + p) * 128;
    const float* iat = init + ((size_t)b * Nn + p) * 64;
    for (int i = threadIdx.x; i < 320; i += 256) {
        float v = (i < 128) ? hat[i]
                 : (i < 256) ? agent[(size_t)row * 128 + (i - 128)]
                             : iat[i - 256];
        cin[i] = v + tvec[TC_OFF + i];
    }
    __syncthreads();
    float s = 0.f;
    for (int i = threadIdx.x; i < 320; i += 256) s += cin[i];
    float mean = block_sum<256>(s, red) / 320.f;
    float s2 = 0.f;
    for (int i = threadIdx.x; i < 320; i += 256) { float d = cin[i] - mean; s2 += d * d; }
    float var = block_sum<256>(s2, red) / 320.f;
    float rstd = rsqrtf(var + 1e-5f);
    for (int i = threadIdx.x; i < 320; i += 256)
        cin[i] = (cin[i] - mean) * rstd * g[i] + bln[i];
    __syncthreads();
    for (int o = threadIdx.x; o < 512; o += 256) {
        float acc = b1[o];
        for (int i = 0; i < 320; ++i) acc += cin[i] * w1[i * 512 + o];
        hid[o] = (acc > 0.f) ? acc : 0.01f * acc;
    }
    __syncthreads();
    if (threadIdx.x < 128) {
        int o = threadIdx.x;
        float u = b2[o], gg = b2[o + 128];
        for (int i = 0; i < 512; ++i) {
            float hv = hid[i];
            u  += hv * w2[i * 256 + o];
            gg += hv * w2[i * 256 + o + 128];
        }
        float sg = 1.f / (1.f + expf(-gg));
        float av = agent[(size_t)row * 128 + o];
        convout[(size_t)row * 128 + o] = sg * av + (1.f - sg) * u;
    }
}

// ---------------------------------------------------------------------------
// agent update (gated), AROWS blocks x 256
// ---------------------------------------------------------------------------
__global__ __launch_bounds__(256) void agent_kernel(
    const float* __restrict__ h, const float* __restrict__ conv,
    const float* __restrict__ glob, const int* __restrict__ pos,
    const float* tvec, const float* g, const float* bln,
    const float* w1, const float* b1, const float* w2, const float* b2,
    float* __restrict__ agent)
{
    int row = blockIdx.x;
    int b = row >> 6;
    int p = pos[row];
    __shared__ float cin[384], hid[512], red[256];
    const float* hat = h + ((size_t)b * Nn + p) * 128;
    for (int i = threadIdx.x; i < 384; i += 256) {
        float v = (i < 128) ? conv[(size_t)row * 128 + i]
                 : (i < 256) ? hat[i - 128]
                             : glob[(size_t)b * 128 + (i - 256)];
        cin[i] = v + tvec[TA_OFF + i];
    }
    __syncthreads();
    float s = 0.f;
    for (int i = threadIdx.x; i < 384; i += 256) s += cin[i];
    float mean = block_sum<256>(s, red) / 384.f;
    float s2 = 0.f;
    for (int i = threadIdx.x; i < 384; i += 256) { float d = cin[i] - mean; s2 += d * d; }
    float var = block_sum<256>(s2, red) / 384.f;
    float rstd = rsqrtf(var + 1e-5f);
    for (int i = threadIdx.x; i < 384; i += 256)
        cin[i] = (cin[i] - mean) * rstd * g[i] + bln[i];
    __syncthreads();
    for (int o = threadIdx.x; o < 512; o += 256) {
        float acc = b1[o];
        for (int i = 0; i < 384; ++i) acc += cin[i] * w1[i * 512 + o];
        hid[o] = (acc > 0.f) ? acc : 0.01f * acc;
    }
    __syncthreads();
    if (threadIdx.x < 128) {
        int o = threadIdx.x;
        float u = b2[o], gg = b2[o + 128];
        for (int i = 0; i < 512; ++i) {
            float hv = hid[i];
            u  += hv * w2[i * 256 + o];
            gg += hv * w2[i * 256 + o + 128];
        }
        float sg = 1.f / (1.f + expf(-gg));
        float cv = conv[(size_t)row * 128 + o];
        agent[(size_t)row * 128 + o] = sg * cv + (1.f - sg) * u;
    }
}

// q16 = f16(lin(ln(agent, query_ln), query_l))   (AROWS blocks x 128)
__global__ __launch_bounds__(128) void query_kernel(
    const float* __restrict__ agent, const float* g, const float* bln,
    const float* w, const float* bw, half_t* __restrict__ q16)
{
    int row = blockIdx.x, tid = threadIdx.x;
    __shared__ float x[128], red[128];
    float v = agent[(size_t)row * 128 + tid];
    float mean = block_sum<128>(v, red) / 128.f;
    float d = v - mean;
    float var = block_sum<128>(d * d, red) / 128.f;
    float rstd = rsqrtf(var + 1e-5f);
    x[tid] = d * rstd * g[tid] + bln[tid];
    __syncthreads();
    float acc = bw[tid];
    for (int i = 0; i < 128; ++i) acc += x[i] * w[i * 128 + tid];
    q16[(size_t)row * 128 + tid] = (half_t)acc;
}

// kln = f16(LN([h|init], key1_ln))   (ROWS blocks x 256)
__global__ __launch_bounds__(256) void ln_key_kernel(
    const float* __restrict__ h, const float* __restrict__ init,
    const float* g, const float* bln, half_t* __restrict__ out)
{
    int row = blockIdx.x;
    __shared__ float sh[192], red[256];
    for (int i = threadIdx.x; i < 192; i += 256)
        sh[i] = (i < 128) ? h[(size_t)row * 128 + i]
                          : init[(size_t)row * 64 + (i - 128)];
    __syncthreads();
    float s = (threadIdx.x < 192) ? sh[threadIdx.x] : 0.f;
    float mean = block_sum<256>(s, red) / 192.f;
    float s2 = 0.f;
    if (threadIdx.x < 192) { float d = sh[threadIdx.x] - mean; s2 = d * d; }
    float var = block_sum<256>(s2, red) / 192.f;
    float rstd = rsqrtf(var + 1e-5f);
    for (int i = threadIdx.x; i < 192; i += 256)
        out[(size_t)row * 192 + i] = (half_t)((sh[i] - mean) * rstd * g[i] + bln[i]);
}

// rln = f16(LN(h + t_read, read_ln))   (ROWS blocks x 128)
__global__ __launch_bounds__(128) void ln_read_kernel(
    const float* __restrict__ h, const float* tvec,
    const float* g, const float* bln, half_t* __restrict__ out)
{
    int row = blockIdx.x, tid = threadIdx.x;
    __shared__ float red[128];
    float v = h[(size_t)row * 128 + tid] + tvec[TR_OFF + tid];
    float mean = block_sum<128>(v, red) / 128.f;
    float d = v - mean;
    float var = block_sum<128>(d * d, red) / 128.f;
    float rstd = rsqrtf(var + 1e-5f);
    out[(size_t)row * 128 + tid] = (half_t)(d * rstd * g[tid] + bln[tid]);
}

// ---------------------------------------------------------------------------
// softmax / argmax / state update per (b,a) row.  logits holds raw dot/sqrt(D).
// ---------------------------------------------------------------------------
__global__ __launch_bounds__(256) void softmax_kernel(
    float* __restrict__ logits, float* __restrict__ vis,
    int* __restrict__ pos, int* __restrict__ prev, float* __restrict__ attn_val,
    const float* attn_w, const float* attn_b,
    const float* back_p, const float* stay_p,
    const float* expl_p, const float* unexp_p)
{
    int row = blockIdx.x, tid = threadIdx.x;
    float aw = attn_w[0], ab = attn_b[0];
    float back = back_p[0], stay = stay_p[0];
    float expl = expl_p[0], unexp = unexp_p[0];
    float* lg = logits + (size_t)row * Nn;
    float* vr = vis    + (size_t)row * Nn;
    int p = pos[row], pp = prev[row];

    __shared__ float redf[256]; __shared__ int redi[256];
    __shared__ float s_max, s_sum; __shared__ int s_idx;

    float lmax = -3.4e38f; int lidx = Nn;
    for (int n = tid; n < Nn; n += 256) {
        float vv = vr[n];
        float v = lg[n] * aw + ab + expl * vv + unexp * (1.f - vv);
        if (n == pp) v += back;
        if (n == p)  v += stay;
        lg[n] = v;
        if (v > lmax) { lmax = v; lidx = n; }
    }
    redf[tid] = lmax; redi[tid] = lidx; __syncthreads();
    for (int o = 128; o > 0; o >>= 1) {
        if (tid < o) {
            float v2 = redf[tid + o]; int i2 = redi[tid + o];
            if (v2 > redf[tid] || (v2 == redf[tid] && i2 < redi[tid])) {
                redf[tid] = v2; redi[tid] = i2;
            }
        }
        __syncthreads();
    }
    if (tid == 0) { s_max = redf[0]; s_idx = redi[0]; }
    __syncthreads();
    float bmax = s_max;
    float s = 0.f;
    for (int n = tid; n < Nn; n += 256) s += expf(lg[n] - bmax);
    float total = block_sum<256>(s, redf);
    if (tid == 0) s_sum = total;
    __syncthreads();
    int newpos = s_idx;
    for (int n = tid; n < Nn; n += 256)
        vr[n] = (n == newpos) ? 1.f : vr[n] * 0.9f;
    if (tid == 0) {
        prev[row] = p;
        pos[row] = newpos;
        attn_val[row] = 1.f / s_sum;   // exp(max - max) / sum
    }
}

// ---------------------------------------------------------------------------
// host orchestration
// ---------------------------------------------------------------------------
// big variant: 2x4 tiles/wave, block tile 64M x 256N (needs M%64==0, N%256==0)
static inline void launch_gemm_big(hipStream_t s,
    const half_t* A, const half_t* Bt, const float* bias,
    float* Cf, half_t* Ch, int M, int N, int K,
    long long sA, long long sB, long long sC, int batches,
    float alpha, float beta, int act)
{
    dim3 g(N / 256, M / 64, batches);
    gemm_wmma_kernel<2, 4><<<g, 256, 0, s>>>(A, Bt, bias, Cf, Ch, M, N, K,
                                             sA, sB, sC, alpha, beta, act);
}
// small variant: 1x1 tile/wave, block tile 32M x 64N
static inline void launch_gemm_small(hipStream_t s,
    const half_t* A, const half_t* Bt, const float* bias,
    float* Cf, half_t* Ch, int M, int N, int K,
    long long sA, long long sB, long long sC, int batches,
    float alpha, float beta, int act)
{
    dim3 g(N / 64, M / 32, batches);
    gemm_wmma_kernel<1, 1><<<g, 256, 0, s>>>(A, Bt, bias, Cf, Ch, M, N, K,
                                             sA, sB, sC, alpha, beta, act);
}

extern "C" void kernel_launch(void* const* d_in, const int* in_sizes, int n_in,
                              void* d_out, int out_size, void* d_ws, size_t ws_size,
                              hipStream_t stream) {
    int ix = 0;
    const float* node_emb   = (const float*)d_in[ix++];   // [B,N,I]
    const int*   start_pos  = (const int*)  d_in[ix++];   // [B]
    ix++;                                                 // num_steps (use 8)
    const float* time_table = (const float*)d_in[ix++];
    const float* time1_w = (const float*)d_in[ix++]; const float* time1_b = (const float*)d_in[ix++];
    const float* time2_w = (const float*)d_in[ix++]; const float* time2_b = (const float*)d_in[ix++];
    const float* inp1_w  = (const float*)d_in[ix++]; const float* inp1_b  = (const float*)d_in[ix++];
    const float* inp2_w  = (const float*)d_in[ix++]; const float* inp2_b  = (const float*)d_in[ix++];
    const float* agent_tab = (const float*)d_in[ix++];
    const float* msgln_g = (const float*)d_in[ix++]; const float* msgln_b = (const float*)d_in[ix++];
    const float* msg_w   = (const float*)d_in[ix++]; const float* msg_b   = (const float*)d_in[ix++];
    const float* ndln_g  = (const float*)d_in[ix++]; const float* ndln_b  = (const float*)d_in[ix++];
    const float* nd1_w   = (const float*)d_in[ix++]; const float* nd1_b   = (const float*)d_in[ix++];
    const float* nd2_w   = (const float*)d_in[ix++]; const float* nd2_b   = (const float*)d_in[ix++];
    const float* agln_g  = (const float*)d_in[ix++]; const float* agln_b  = (const float*)d_in[ix++];
    const float* ag1_w   = (const float*)d_in[ix++]; const float* ag1_b   = (const float*)d_in[ix++];
    const float* ag2_w   = (const float*)d_in[ix++]; const float* ag2_b   = (const float*)d_in[ix++];
    const float* cvln_g  = (const float*)d_in[ix++]; const float* cvln_b  = (const float*)d_in[ix++];
    const float* cv1_w   = (const float*)d_in[ix++]; const float* cv1_b   = (const float*)d_in[ix++];
    const float* cv2_w   = (const float*)d_in[ix++]; const float* cv2_b   = (const float*)d_in[ix++];
    const float* glln_g  = (const float*)d_in[ix++]; const float* glln_b  = (const float*)d_in[ix++];
    const float* gl1_w   = (const float*)d_in[ix++]; const float* gl1_b   = (const float*)d_in[ix++];
    const float* gl2_w   = (const float*)d_in[ix++]; const float* gl2_b   = (const float*)d_in[ix++];
    const float* rdln_g  = (const float*)d_in[ix++]; const float* rdln_b  = (const float*)d_in[ix++];
    const float* rd_w    = (const float*)d_in[ix++]; const float* rd_b    = (const float*)d_in[ix++];
    const float* out_w   = (const float*)d_in[ix++]; const float* out_b   = (const float*)d_in[ix++];
    const float* tn_w    = (const float*)d_in[ix++]; const float* tn_b    = (const float*)d_in[ix++];
    const float* ta_w    = (const float*)d_in[ix++]; const float* ta_b    = (const float*)d_in[ix++];
    const float* tc_w    = (const float*)d_in[ix++]; const float* tc_b    = (const float*)d_in[ix++];
    const float* tg_w    = (const float*)d_in[ix++]; const float* tg_b    = (const float*)d_in[ix++];
    const float* tr_w    = (const float*)d_in[ix++]; const float* tr_b    = (const float*)d_in[ix++];
    const float* k1ln_g  = (const float*)d_in[ix++]; const float* k1ln_b  = (const float*)d_in[ix++];
    const float* k1_w    = (const float*)d_in[ix++]; const float* k1_b    = (const float*)d_in[ix++];
    const float* qln_g   = (const float*)d_in[ix++]; const float* qln_b   = (const float*)d_in[ix++];
    const float* q_w     = (const float*)d_in[ix++]; const float* q_b     = (const float*)d_in[ix++];
    const float* attn_w  = (const float*)d_in[ix++]; const float* attn_bb = (const float*)d_in[ix++];
    const float* back_p  = (const float*)d_in[ix++]; const float* stay_p  = (const float*)d_in[ix++];
    const float* expl_p  = (const float*)d_in[ix++]; const float* unexp_p = (const float*)d_in[ix++];
    (void)n_in; (void)in_sizes; (void)out_size;

    // ------------------- workspace carve-up -------------------
    char* base = (char*)d_ws;
    size_t cur = 0;
    auto alloc = [&](size_t bytes) -> void* {
        void* p = base + cur;
        cur = (cur + bytes + 255) & ~(size_t)255;
        return p;
    };
    float*  h      = (float*) alloc((size_t)ROWS * Dd * 4);
    float*  agg    = (float*) alloc((size_t)ROWS * Dd * 4);
    half_t* emb16  = (half_t*)alloc((size_t)ROWS * Ii * 2);
    half_t* nln16  = (half_t*)alloc((size_t)ROWS * 448 * 2);
    half_t* hid16  = (half_t*)alloc((size_t)ROWS * 512 * 2);
    float*  ug     = (float*) alloc((size_t)ROWS * 256 * 4);
    half_t* kln16  = (half_t*)alloc((size_t)ROWS * 192 * 2);
    half_t* k16    = (half_t*)alloc((size_t)ROWS * Dd * 2);
    half_t* q16    = (half_t*)alloc((size_t)AROWS * Dd * 2);
    float*  vis    = (float*) alloc((size_t)AROWS * Nn * 4);
    float*  agent  = (float*) alloc((size_t)AROWS * Dd * 4);
    float*  conv   = (float*) alloc((size_t)AROWS * Dd * 4);
    float*  msg    = (float*) alloc((size_t)AROWS * Dd * 4);
    float*  glob   = (float*) alloc((size_t)Bsz * Dd * 4);
    float*  attnv  = (float*) alloc((size_t)AROWS * 4);
    int*    pos    = (int*)   alloc((size_t)AROWS * 4);
    int*    prev   = (int*)   alloc((size_t)AROWS * 4);
    float*  tvec   = (float*) alloc((size_t)TVEC_LEN * 4);
    half_t* wt_inp1 = (half_t*)alloc(256 * 64 * 2);
    half_t* wt_inp2 = (half_t*)alloc(128 * 256 * 2);
    half_t* wt_nd1  = (half_t*)alloc(512 * 448 * 2);
    half_t* wt_nd2  = (half_t*)alloc(256 * 512 * 2);
    half_t* wt_key  = (half_t*)alloc(128 * 192 * 2);
    half_t* wt_read = (half_t*)alloc(128 * 128 * 2);
    half_t* wt_out  = (half_t*)alloc(64 * 128 * 2);
    // aliases (lifetimes don't overlap):
    float*  logits  = ug;               // [B,A,N] = 16.8MB <= ug 67MB
    half_t* rln16   = nln16;            // readout LN, after loop
    half_t* final16 = kln16;            // readout hidden, after loop
    if (ws_size < cur) return;          // insufficient scratch: no-op

    float* outp = (float*)d_out;

    // ------------------- one-time prep -------------------
    auto wtrans = [&](const float* w, half_t* wt, int fin, int fout) {
        long long tot = (long long)fin * fout;
        wtrans_kernel<<<(unsigned)((tot + 255) / 256), 256, 0, stream>>>(w, wt, fin, fout);
    };
    wtrans(inp1_w, wt_inp1, 64, 256);
    wtrans(inp2_w, wt_inp2, 256, 128);
    wtrans(nd1_w,  wt_nd1,  448, 512);
    wtrans(nd2_w,  wt_nd2,  512, 256);
    wtrans(k1_w,   wt_key,  192, 128);
    wtrans(rd_w,   wt_read, 128, 128);
    wtrans(out_w,  wt_out,  128, 64);

    {
        long long n = (long long)ROWS * Ii;
        convert_h_kernel<<<(unsigned)((n + 255) / 256), 256, 0, stream>>>(node_emb, emb16, n);
    }
    // input projection: h = lin(lrelu(lin(emb, inp_l1)), inp_l2)
    launch_gemm_big(stream, emb16, wt_inp1, inp1_b, nullptr, hid16,
                    ROWS, 256, 64, 0, 0, 0, 1, 1.f, 0.f, 1);
    launch_gemm_small(stream, hid16, wt_inp2, inp2_b, h, nullptr,
                      ROWS, 128, 256, 0, 0, 0, 1, 1.f, 0.f, 0);

    init_agent_kernel<<<(AROWS * Dd + 255) / 256, 256, 0, stream>>>(agent_tab, agent);
    init_state_kernel<<<AROWS, 256, 0, stream>>>(start_pos, vis, pos, prev, attnv);

    const float inv_sqrt_d = 0.088388347648318447f;   // 1/sqrt(128)

    // ------------------- time steps -------------------
    for (int t = 1; t <= NSTEPS; ++t) {
        time_kernel<<<1, 512, 0, stream>>>(time_table, t,
            time1_w, time1_b, time2_w, time2_b,
            tn_w, tn_b, ta_w, ta_b, tc_w, tc_b, tg_w, tg_b, tr_w, tr_b, tvec);

        msg_kernel<<<AROWS, 128, 0, stream>>>(agent, msgln_g, msgln_b, msg_w, msg_b, msg);

        {
            long long n = (long long)ROWS * Dd;
            fillf_kernel<<<(unsigned)((n + 255) / 256), 256, 0, stream>>>(agg, 0.f, n);
        }
        scatter_kernel<<<AROWS, 128, 0, stream>>>(msg, attnv, pos, agg);

        glob_kernel<<<Bsz, 128, 0, stream>>>(agent, tvec, glln_g, glln_b,
                                             gl1_w, gl1_b, gl2_w, gl2_b, glob);

        ln_node_kernel<<<ROWS, 256, 0, stream>>>(h, agg, glob, node_emb, tvec,
                                                 ndln_g, ndln_b, nln16);
        launch_gemm_big(stream, nln16, wt_nd1, nd1_b, nullptr, hid16,
                        ROWS, 512, 448, 0, 0, 0, 1, 1.f, 0.f, 1);
        launch_gemm_big(stream, hid16, wt_nd2, nd2_b, ug, nullptr,
                        ROWS, 256, 512, 0, 0, 0, 1, 1.f, 0.f, 0);
        {
            long long n = (long long)ROWS * Dd;
            gate_h_kernel<<<(unsigned)((n + 255) / 256), 256, 0, stream>>>(ug, h);
        }

        conv_kernel<<<AROWS, 256, 0, stream>>>(h, agent, node_emb, pos, tvec,
            cvln_g, cvln_b, cv1_w, cv1_b, cv2_w, cv2_b, conv);
        agent_kernel<<<AROWS, 256, 0, stream>>>(h, conv, glob, pos, tvec,
            agln_g, agln_b, ag1_w, ag1_b, ag2_w, ag2_b, agent);

        query_kernel<<<AROWS, 128, 0, stream>>>(agent, qln_g, qln_b, q_w, q_b, q16);
        ln_key_kernel<<<ROWS, 256, 0, stream>>>(h, node_emb, k1ln_g, k1ln_b, kln16);
        launch_gemm_small(stream, kln16, wt_key, k1_b, nullptr, k16,
                          ROWS, 128, 192, 0, 0, 0, 1, 1.f, 0.f, 0);

        // logits[b] = (q[b] @ k[b]^T) / sqrt(D)    (attn_l affine applied in softmax)
        launch_gemm_big(stream, q16, k16, nullptr, logits, nullptr,
                        Aa, Nn, Dd,
                        (long long)Aa * Dd, (long long)Nn * Dd, (long long)Aa * Nn,
                        Bsz, inv_sqrt_d, 0.f, 0);

        softmax_kernel<<<AROWS, 256, 0, stream>>>(logits, vis, pos, prev, attnv,
            attn_w, attn_bb, back_p, stay_p, expl_p, unexp_p);
    }

    // ------------------- readout (te at t = NSTEPS already in tvec) -------------------
    ln_read_kernel<<<ROWS, 128, 0, stream>>>(h, tvec, rdln_g, rdln_b, rln16);
    launch_gemm_small(stream, rln16, wt_read, rd_b, nullptr, final16,
                      ROWS, 128, 128, 0, 0, 0, 1, 1.f, 0.f, 1);
    launch_gemm_small(stream, final16, wt_out, out_b, outp, nullptr,
                      ROWS, 64, 128, 0, 0, 0, 1, 1.f, 0.f, 0);
}
